// Detector_77979426226960
// MI455X (gfx1250) — matrix operand order repaired and checked by reading, not compiled, not run on hardware
//
#include <hip/hip_runtime.h>
#include <stdint.h>

// ---------------- CDNA5 WMMA types ----------------
typedef __attribute__((ext_vector_type(16))) __bf16 bf16x16;
typedef __attribute__((ext_vector_type(8)))  float  floatx8;

#define TT   32
#define NN   256
#define HH   4
#define EMB  32
#define CIN  32
#define ORIG 32

#define ACT_NONE 0
#define ACT_ELU  1
#define ACT_TANH 2

__device__ __forceinline__ float act_apply(float v, int act) {
  if (act == ACT_ELU)  return v > 0.f ? v : (__expf(v) - 1.f);
  if (act == ACT_TANH) return tanhf(v);
  return v;
}

__device__ __forceinline__ void fill8(bf16x16& f, int base, float4 u, float4 v) {
  f[base + 0] = (__bf16)u.x; f[base + 1] = (__bf16)u.y;
  f[base + 2] = (__bf16)u.z; f[base + 3] = (__bf16)u.w;
  f[base + 4] = (__bf16)v.x; f[base + 5] = (__bf16)v.y;
  f[base + 6] = (__bf16)v.z; f[base + 7] = (__bf16)v.w;
}

// =====================================================================
// Batched GEMM: C[b] = act(alpha * A[b] @ B[b] + bias)
// EXACT TILING ASSUMED: M % 32 == 0, Nout % 32 == 0, K % 32 == 0,
// lda % 4 == 0 and 16B-aligned A (true for all callers).
// One wave computes a 32x32 C tile: 2 A frags x 2 B frags -> 4 WMMA
// per K-step (v_wmma_f32_16x16x32_bf16, bf16 in / f32 acc).
// A fragments use explicit float4 loads (global_load_b128).
// =====================================================================
__global__ __launch_bounds__(32)
void k_gemm(const float* __restrict__ A, long long sAb, int lda,
            const float* __restrict__ B, long long sBb, int ldbr, int ldbc,
            const float* __restrict__ bias,
            float* __restrict__ C, long long sCb, int ldc,
            int K, float alpha, int act)
{
  const int tm = blockIdx.x * 32;
  const int tn = blockIdx.y * 32;
  const int b  = blockIdx.z;
  A += (long long)b * sAb;
  B += (long long)b * sBb;
  C += (long long)b * sCb;

  const int lane = threadIdx.x & 31;
  const int half = lane >> 4;   // K-half selector (wave32 WMMA layout)
  const int l16  = lane & 15;

  floatx8 acc00 = {}, acc01 = {}, acc10 = {}, acc11 = {};
  // A: lane row; element e<8 -> k0+half*8+e ; e>=8 -> k0+16+half*8+(e-8)
  const float* Arow0 = A + (long long)(tm + l16) * lda + (half << 3);
  const float* Arow1 = Arow0 + (long long)16 * lda;
  // B: lane column; element e<8 -> k0+half*8+e ; e>=8 -> k0+16+half*8+(e-8)
  const float* Bc0 = B + (long long)(tn + l16) * ldbc + (long long)(half << 3) * ldbr;
  const float* Bc1 = Bc0 + (long long)16 * ldbc;

  for (int k0 = 0; k0 < K; k0 += 32) {
    __builtin_prefetch(&Arow0[k0 + 32], 0, 1);   // global_prefetch_b8
    __builtin_prefetch(&Arow1[k0 + 32], 0, 1);

    const float4* pa0 = (const float4*)(Arow0 + k0);
    const float4* pa1 = (const float4*)(Arow1 + k0);
    float4 a00 = pa0[0], a01 = pa0[1], a02 = pa0[4], a03 = pa0[5];
    float4 a10 = pa1[0], a11 = pa1[1], a12 = pa1[4], a13 = pa1[5];

    bf16x16 af0, af1, bf0, bf1;
    fill8(af0, 0, a00, a01); fill8(af0, 8, a02, a03);
    fill8(af1, 0, a10, a11); fill8(af1, 8, a12, a13);
    #pragma unroll
    for (int e = 0; e < 16; ++e) {
      long long o = (long long)(k0 + e + (e >= 8 ? 8 : 0)) * ldbr;
      bf0[e] = (__bf16)Bc0[o];
      bf1[e] = (__bf16)Bc1[o];
    }
    acc00 = __builtin_amdgcn_wmma_f32_16x16x32_bf16(
        false, af0, false, bf0, (short)0, acc00, false, false);
    acc01 = __builtin_amdgcn_wmma_f32_16x16x32_bf16(
        false, af0, false, bf1, (short)0, acc01, false, false);
    acc10 = __builtin_amdgcn_wmma_f32_16x16x32_bf16(
        false, af1, false, bf0, (short)0, acc10, false, false);
    acc11 = __builtin_amdgcn_wmma_f32_16x16x32_bf16(
        false, af1, false, bf1, (short)0, acc11, false, false);
  }

  const int cn0 = tn + l16, cn1 = cn0 + 16;
  const float bv0 = bias ? bias[cn0] : 0.0f;
  const float bv1 = bias ? bias[cn1] : 0.0f;
  #pragma unroll
  for (int r = 0; r < 8; ++r) {
    const int cm0 = tm + r + (half << 3);
    const int cm1 = cm0 + 16;
    C[(long long)cm0 * ldc + cn0] = act_apply(alpha * acc00[r] + bv0, act);
    C[(long long)cm0 * ldc + cn1] = act_apply(alpha * acc01[r] + bv1, act);
    C[(long long)cm1 * ldc + cn0] = act_apply(alpha * acc10[r] + bv0, act);
    C[(long long)cm1 * ldc + cn1] = act_apply(alpha * acc11[r] + bv1, act);
  }
}

// =====================================================================
// GATv2 attention logits + masked softmax over sources j.
// xl/xr layout: [j][h][f] contiguous ((j*H+h)*F+f). One wave per (i,h).
// Output a[i, j*H + h] (head-mean aggregation becomes ONE GEMM with
// K = N*H against B = xl viewed as [(j,h), f], ldb = F).
// F is a template parameter (32/64/128): unrolled, float4 loads.
// =====================================================================
template<int F>
__global__ __launch_bounds__(32)
void k_attn(const float* __restrict__ xl, const float* __restrict__ xr,
            long long sXb, const float* __restrict__ att,
            const unsigned char* __restrict__ mask, long long sMb,
            float* __restrict__ aout, long long sAb)
{
  const int i = blockIdx.x, h = blockIdx.y, b = blockIdx.z;
  xl   += (long long)b * sXb;
  xr   += (long long)b * sXb;
  mask += (long long)b * sMb;
  aout += (long long)b * sAb;
  const int lane = threadIdx.x;

  __shared__ float s_xr[F];
  __shared__ float s_att[F];
  #pragma unroll
  for (int f = lane; f < F; f += 32) {
    s_xr[f]  = xr[((long long)i * HH + h) * F + f];
    s_att[f] = att[h * F + f];
  }
  __syncthreads();

  float e[8];
  #pragma unroll
  for (int jc = 0; jc < 8; ++jc) {
    const int j = jc * 32 + lane;
    const float4* xlj = (const float4*)(xl + ((long long)j * HH + h) * F);
    float acc = 0.f;
    #pragma unroll
    for (int q = 0; q < F / 4; ++q) {
      float4 v = xlj[q];
      float z0 = s_xr[4*q+0] + v.x; z0 = z0 > 0.f ? z0 : 0.2f * z0;
      float z1 = s_xr[4*q+1] + v.y; z1 = z1 > 0.f ? z1 : 0.2f * z1;
      float z2 = s_xr[4*q+2] + v.z; z2 = z2 > 0.f ? z2 : 0.2f * z2;
      float z3 = s_xr[4*q+3] + v.w; z3 = z3 > 0.f ? z3 : 0.2f * z3;
      acc += s_att[4*q+0]*z0 + s_att[4*q+1]*z1 + s_att[4*q+2]*z2 + s_att[4*q+3]*z3;
    }
    e[jc] = mask[(long long)i * NN + j] ? acc : -1e30f;
  }
  float m = -1e30f;
  #pragma unroll
  for (int jc = 0; jc < 8; ++jc) m = fmaxf(m, e[jc]);
  for (int off = 16; off; off >>= 1) m = fmaxf(m, __shfl_xor(m, off, 32));
  float s = 0.f, ex[8];
  #pragma unroll
  for (int jc = 0; jc < 8; ++jc) { ex[jc] = __expf(e[jc] - m); s += ex[jc]; }
  for (int off = 16; off; off >>= 1) s += __shfl_xor(s, off, 32);
  const float inv = 1.0f / s;
  #pragma unroll
  for (int jc = 0; jc < 8; ++jc) {
    const int j = jc * 32 + lane;
    aout[(long long)i * (NN * HH) + (long long)j * HH + h] = ex[jc] * inv;
  }
}

// ---------------- small utility kernels ----------------
__global__ void k_copy_act(const float* __restrict__ src, long long sSb, int lds_,
                           float* __restrict__ dst, long long sDb, int ldd,
                           long long rows, int cols, int act)
{
  long long idx = (long long)blockIdx.x * 256 + threadIdx.x;
  int b = blockIdx.y;
  if (idx >= rows * cols) return;
  long long r = idx / cols; int c = (int)(idx % cols);
  float v = src[(long long)b * sSb + r * lds_ + c];
  dst[(long long)b * sDb + r * ldd + c] = act_apply(v, act);
}

__global__ void k_build_mask(const unsigned char* __restrict__ edge,
                             unsigned char* __restrict__ mask)
{
  long long idx = (long long)blockIdx.x * 256 + threadIdx.x;
  if (idx >= (long long)TT * NN * NN) return;
  long long t = idx / (NN * NN);
  int rem = (int)(idx % (NN * NN));
  int i = rem / NN, j = rem % NN;
  mask[idx] = (unsigned char)((edge[t * NN * NN + (long long)j * NN + i] != 0) || (i == j));
}

__global__ void k_concat_emb(const float* __restrict__ emb, const int* __restrict__ sidx,
                             float* __restrict__ xf)
{
  long long idx = (long long)blockIdx.x * 256 + threadIdx.x;
  if (idx >= (long long)TT * NN * EMB) return;
  long long tn = idx / EMB; int c = (int)(idx % EMB);
  int n = (int)(tn % NN);
  xf[tn * 64 + CIN + c] = emb[(long long)sidx[n] * EMB + c];
}

__global__ void k_tanh_pair(const float* __restrict__ base, const float* __restrict__ full,
                            float* __restrict__ hcom, float* __restrict__ hself, int n)
{
  int idx = blockIdx.x * 256 + threadIdx.x;
  if (idx >= n) return;
  hcom[idx]  = tanhf(base[idx] + full[idx]);
  hself[idx] = tanhf(base[idx]);
}

// pm[s,j,:] = (j == b0+s) ? pmB[j,:] : pmA[j,:]
__global__ void k_build_pm(float* __restrict__ pm, const float* __restrict__ pmA,
                           const float* __restrict__ pmB, int b0)
{
  long long idx = (long long)blockIdx.x * 256 + threadIdx.x;
  if (idx >= (long long)32 * NN * 32) return;
  int s = (int)(idx / (NN * 32));
  int rem = (int)(idx % (NN * 32));
  int j = rem / 32, c = rem % 32;
  pm[idx] = (j == b0 + s) ? pmB[j * 32 + c] : pmA[j * 32 + c];
}

// gtmp[s,:] = g[s, b0+s, :]
__global__ void k_gather_rows(float* __restrict__ gtmp, const float* __restrict__ g, int b0)
{
  int idx = blockIdx.x * 256 + threadIdx.x;
  if (idx >= 32 * 32) return;
  int s = idx / 32, c = idx % 32;
  gtmp[idx] = g[(long long)s * NN * 32 + (long long)(b0 + s) * 32 + c];
}

// =====================================================================
// Host orchestration
// =====================================================================
struct GatP { const float *Wl, *Wr, *att, *b, *bl, *br; int F; };

extern "C" void kernel_launch(void* const* d_in, const int* in_sizes, int n_in,
                              void* d_out, int out_size, void* d_ws, size_t ws_size,
                              hipStream_t stream)
{
  (void)in_sizes; (void)n_in; (void)out_size; (void)ws_size;

  // ---- input pointers (pytree flattened alphabetically) ----
  const float* x    = (const float*)d_in[0];                 // [T,N,32]
  const unsigned char* edge = (const unsigned char*)d_in[1]; // [T,N,N] bool
  const int*   sidx = (const int*)d_in[2];                   // [N]
  auto F32 = [&](int i) { return (const float*)d_in[i]; };
  auto GAT = [&](int base, int F) {
    return GatP{ F32(base), F32(base + 1), F32(base + 2),
                 F32(base + 3), F32(base + 4), F32(base + 5), F };
  };
  // gd: conv1..4, f11, f2, f3_b, f3_w, f4_b, f4_w, r11, r12, r2, r3_b, r3_w, r4_b, r4_w
  GatP conv1 = GAT(3, 128), conv2 = GAT(9, 64), conv3 = GAT(15, 64), conv4 = GAT(21, 64);
  GatP f11 = GAT(27, 64), f2p = GAT(33, 128);
  const float *f3_b = F32(39), *f3_w = F32(40), *f4_b = F32(41), *f4_w = F32(42);
  GatP r11 = GAT(43, 64), r12 = GAT(49, 64), r2p = GAT(55, 128);
  const float *r3_b = F32(61), *r3_w = F32(62), *r4_b = F32(63), *r4_w = F32(64);
  // nd: aggr1_b, aggr1_w, aggr2_b, aggr2_w, embedding_projection, gc1, gc2,
  //     masked_proj, node_projection, normal_proj, rec_b, rec_w
  const float *aggr1_b = F32(65), *aggr1_w = F32(66), *aggr2_b = F32(67), *aggr2_w = F32(68);
  const float *emb_proj = F32(69);
  GatP gc1 = GAT(70, 32), gc2 = GAT(76, 32);
  const float *masked_proj = F32(82), *node_proj = F32(83), *normal_proj = F32(84);
  const float *rec_b = F32(85), *rec_w = F32(86);
  const float *proj = F32(87), *semb = F32(88);

  float* out = (float*)d_out;   // [recon 8192 | forecast 8192 | node 8192 | E 524288]

  // ---- workspace bump allocator ----
  uintptr_t wbase = (uintptr_t)d_ws;
  size_t off = 0;
  auto alloc = [&](size_t bytes) -> void* {
    off = (off + 255) & ~(size_t)255;
    void* p = (void*)(wbase + off);
    off += bytes;
    return p;
  };
  float* xf    = (float*)alloc((size_t)TT * NN * 64 * 4);
  unsigned char* maskb = (unsigned char*)alloc((size_t)TT * NN * NN);
  float* out1  = (float*)alloc((size_t)TT * NN * 128 * 4);
  float* e1    = (float*)alloc((size_t)TT * NN * 64 * 4);
  float* e2    = (float*)alloc((size_t)TT * NN * 128 * 4);
  float* Ebuf  = (float*)alloc((size_t)TT * NN * 64 * 4);
  float* XL    = (float*)alloc((size_t)TT * NN * 512 * 4);
  float* XR    = (float*)alloc((size_t)TT * NN * 512 * 4);
  float* AB    = (float*)alloc((size_t)TT * NN * NN * HH * 4);
  float* hs1   = (float*)alloc((size_t)NN * 64 * 4);
  float* hs2   = (float*)alloc((size_t)NN * 128 * 4);
  float* hs3   = (float*)alloc((size_t)NN * 128 * 4);
  float* hs4   = (float*)alloc((size_t)NN * 64 * 4);
  float* xp    = (float*)alloc((size_t)NN * 64 * 4);
  float* ep    = (float*)alloc((size_t)NN * 64 * 4);
  float* baseb = (float*)alloc((size_t)NN * 64 * 4);
  float* fullb = (float*)alloc((size_t)NN * 64 * 4);
  float* hcom  = (float*)alloc((size_t)NN * 64 * 4);
  float* hself = (float*)alloc((size_t)NN * 64 * 4);
  float* h2c   = (float*)alloc((size_t)NN * 32 * 4);
  float* h2s   = (float*)alloc((size_t)NN * 32 * 4);
  float* pmA   = (float*)alloc((size_t)NN * 32 * 4);
  float* pmB   = (float*)alloc((size_t)NN * 32 * 4);
  float* pm    = (float*)alloc((size_t)32 * NN * 32 * 4);
  float* g1    = (float*)alloc((size_t)32 * NN * 32 * 4);
  float* g2    = (float*)alloc((size_t)32 * NN * 32 * 4);
  float* gtmp  = (float*)alloc((size_t)32 * 32 * 4);

  // ---- launch helpers ----
  auto gemm = [&](const float* A, long long sAb, int lda,
                  const float* B, long long sBb, int ldbr, int ldbc,
                  const float* bias, float* C, long long sCb, int ldc,
                  int M, int K, int Nout, float alpha, int act, int batch) {
    dim3 g(M / 32, Nout / 32, batch);   // exact tiling guaranteed by callers
    k_gemm<<<g, dim3(32), 0, stream>>>(A, sAb, lda, B, sBb, ldbr, ldbc, bias,
                                       C, sCb, ldc, K, alpha, act);
  };
  auto copy_act = [&](const float* s, long long sSb, int lds_, float* d, long long sDb,
                      int ldd, long long rows, int cols, int act, int batch) {
    dim3 g((unsigned)((rows * cols + 255) / 256), batch);
    k_copy_act<<<g, 256, 0, stream>>>(s, sSb, lds_, d, sDb, ldd, rows, cols, act);
  };
  auto attn = [&](const float* xlp, const float* xrp, long long sXb, const float* attw,
                  const unsigned char* maskp, long long sMb, int F, int batch) {
    dim3 ga(NN, HH, batch);
    if (F == 32)
      k_attn<32><<<ga, 32, 0, stream>>>(xlp, xrp, sXb, attw, maskp, sMb,
                                        AB, (long long)NN * NN * HH);
    else if (F == 64)
      k_attn<64><<<ga, 32, 0, stream>>>(xlp, xrp, sXb, attw, maskp, sMb,
                                        AB, (long long)NN * NN * HH);
    else
      k_attn<128><<<ga, 32, 0, stream>>>(xlp, xrp, sXb, attw, maskp, sMb,
                                         AB, (long long)NN * NN * HH);
  };
  // full GATv2 (mean over heads + bias + act), batched over grid.z
  auto gatv2 = [&](const float* X, long long sXb, int Fin, GatP p,
                   const unsigned char* maskp, long long sMb,
                   float* outp, long long sOb, int ldco, int act, int batch) {
    int F = p.F, HF = HH * F;
    gemm(X, sXb, Fin, p.Wl, 0, HF, 1, p.bl, XL, (long long)NN * HF, HF,
         NN, Fin, HF, 1.f, ACT_NONE, batch);
    gemm(X, sXb, Fin, p.Wr, 0, HF, 1, p.br, XR, (long long)NN * HF, HF,
         NN, Fin, HF, 1.f, ACT_NONE, batch);
    attn(XL, XR, (long long)NN * HF, p.att, maskp, sMb, F, batch);
    gemm(AB, (long long)NN * NN * HH, NN * HH, XL, (long long)NN * HF, F, 1,
         p.b, outp, sOb, ldco, NN, NN * HH, F, 1.f / HH, act, batch);
  };

  // ================= build inputs =================
  // xf[:,:,0:32] = einsum(tnc,ncd->tnd): per-node 32x32 GEMM, batched over n
  gemm(x, 32, NN * CIN, proj, (long long)CIN * CIN, CIN, 1, nullptr,
       xf, 64, NN * 64, TT, CIN, CIN, 1.f, ACT_NONE, NN);
  {
    long long tot = (long long)TT * NN * EMB;
    k_concat_emb<<<(unsigned)((tot + 255) / 256), 256, 0, stream>>>(semb, sidx, xf);
    long long mt = (long long)TT * NN * NN;
    k_build_mask<<<(unsigned)((mt + 255) / 256), 256, 0, stream>>>(edge, maskb);
  }

  // ================= graph detector (batched over T) =================
  const long long sM = (long long)NN * NN;
  gatv2(xf,   (long long)NN * 64,  64,  conv1, maskb, sM, out1, (long long)NN * 128, 128, ACT_ELU, TT);
  gatv2(out1, (long long)NN * 128, 128, conv2, maskb, sM, e1,   (long long)NN * 64,  64,  ACT_ELU, TT);
  gatv2(e1,   (long long)NN * 64,  64,  conv3, maskb, sM, e2,   (long long)NN * 128, 128, ACT_ELU, TT);
  copy_act(e1, (long long)NN * 64, 64, e2 + 64, (long long)NN * 128, 128, NN, 64, ACT_ELU, TT);
  gatv2(e2,   (long long)NN * 128, 128, conv4, maskb, sM, Ebuf, (long long)NN * 64,  64,  ACT_ELU, TT);

  const float* Elast = Ebuf + (size_t)(TT - 1) * NN * 64;
  const float* Eprev = Ebuf + (size_t)(TT - 2) * NN * 64;
  const unsigned char* mlast = maskb + (size_t)(TT - 1) * NN * NN;
  const unsigned char* mprev = maskb + (size_t)(TT - 2) * NN * NN;

  // ================= recon / forecast heads =================
  auto head = [&](const float* hin, const unsigned char* mk, GatP p11, GatP p12, GatP p2,
                  const float* w3, const float* b3, const float* w4, const float* b4,
                  float* dst) {
    gatv2(hin, 0, 64, p11, mk, 0, hs1, 0, 64, ACT_ELU, 1);
    gatv2(hs1, 0, 64, p12, mk, 0, hs2, 0, 128, ACT_ELU, 1);
    copy_act(hs1, 0, 64, hs2 + 64, 0, 128, NN, 64, ACT_ELU, 1);
    gatv2(hs2, 0, 128, p2, mk, 0, hs3, 0, 128, ACT_ELU, 1);
    gemm(hs3, 0, 128, w3, 0, 64, 1, b3, hs4, 0, 64, NN, 128, 64, 1.f, ACT_TANH, 1);
    gemm(hs4, 0, 64, w4, 0, ORIG, 1, b4, dst, 0, ORIG, NN, 64, ORIG, 1.f, ACT_NONE, 1);
  };
  head(Elast, mlast, r11, r12, r2p, r3_w, r3_b, r4_w, r4_b, out);
  head(Eprev, mprev, f11, r12, f2p, f3_w, f3_b, f4_w, f4_b, out + (size_t)NN * ORIG);

  // ================= node detector =================
  const float* xf_last = xf + (size_t)(TT - 1) * NN * 64;
  gemm(xf_last, 0, 64, node_proj, 0, 64, 1, nullptr, xp, 0, 64, NN, 64, 64, 1.f, ACT_NONE, 1);
  gemm(Eprev,   0, 64, emb_proj,  0, 64, 1, nullptr, ep, 0, 64, NN, 64, 64, 1.f, ACT_NONE, 1);
  // a0 = aggr1_w[:,:,0].T  -> B(k,n) = aggr1_w[n*128 + k*2 + 0]
  gemm(ep, 0, 64, aggr1_w,     0, 2, 128, aggr1_b, baseb, 0, 64, NN, 64, 64, 1.f, ACT_NONE, 1);
  gemm(xp, 0, 64, aggr1_w + 1, 0, 2, 128, nullptr, fullb, 0, 64, NN, 64, 64, 1.f, ACT_NONE, 1);
  k_tanh_pair<<<(NN * 64 + 255) / 256, 256, 0, stream>>>(baseb, fullb, hcom, hself, NN * 64);
  gemm(hcom,  0, 64, aggr2_w, 0, 32, 1, aggr2_b, h2c, 0, 32, NN, 64, 32, 1.f, ACT_NONE, 1);
  gemm(hself, 0, 64, aggr2_w, 0, 32, 1, aggr2_b, h2s, 0, 32, NN, 64, 32, 1.f, ACT_NONE, 1);
  gemm(h2c, 0, 32, normal_proj, 0, 32, 1, nullptr, pmA, 0, 32, NN, 32, 32, 1.f, ACT_NONE, 1);
  gemm(h2s, 0, 32, masked_proj, 0, 32, 1, nullptr, pmB, 0, 32, NN, 32, 32, 1.f, ACT_NONE, 1);

  for (int c = 0; c < NN / 32; ++c) {   // 8 chunks of 32 node-steps
    int b0 = c * 32;
    long long tot = (long long)32 * NN * 32;
    k_build_pm<<<(unsigned)((tot + 255) / 256), 256, 0, stream>>>(pm, pmA, pmB, b0);
    gatv2(pm, (long long)NN * 32, 32, gc1, mlast, 0, g1, (long long)NN * 32, 32, ACT_ELU, 32);
    gatv2(g1, (long long)NN * 32, 32, gc2, mlast, 0, g2, (long long)NN * 32, 32, ACT_ELU, 32);
    k_gather_rows<<<4, 256, 0, stream>>>(gtmp, g2, b0);
    gemm(gtmp, 0, 32, rec_w, 0, ORIG, 1, rec_b,
         out + (size_t)2 * NN * ORIG + (size_t)b0 * ORIG, 0, ORIG,
         32, 32, ORIG, 1.f, ACT_TANH, 1);
  }

  // ================= E output =================
  copy_act(Ebuf, 0, 64, out + (size_t)3 * NN * ORIG, 0, 64,
           (long long)TT * NN, 64, ACT_NONE, 1);
}
